// GCN_67156108640892
// MI455X (gfx1250) — matrix operand order repaired and checked by reading, not compiled, not run on hardware
//
#include <hip/hip_runtime.h>

// ---------------------------------------------------------------------------
// GCN forward on gfx1250 (MI455X):
//   - one fp32 pass over A: row-sums + raw bf16(A+I) copy (saves a 256MB read)
//   - in-place d_i*d_j scaling of the bf16 copy (L2-resident, 128MB < 192MB L2)
//   - all GEMMs: v_wmma_f32_16x16x32_bf16, fp32 accumulate, double-buffered
//     LDS; A tiles moved by the Tensor Data Mover (tensor_load_to_lds, with
//     hardware LDS row padding), completion via TENSORcnt.
// ---------------------------------------------------------------------------

#define NN 8192  // number of graph nodes

typedef __attribute__((ext_vector_type(16))) __bf16 bf16x16;
typedef __attribute__((ext_vector_type(8)))  float  f32x8;
typedef __attribute__((ext_vector_type(4)))  unsigned int u32x4;
typedef __attribute__((ext_vector_type(8)))  int  i32x8;
typedef __attribute__((ext_vector_type(4)))  int  i32x4;

static __device__ __forceinline__ unsigned short f2bf(float f) {
  union { float f; unsigned u; } x; x.f = f;
  unsigned r = x.u + 0x7FFFu + ((x.u >> 16) & 1u);  // round-to-nearest-even
  return (unsigned short)(r >> 16);
}
static __device__ __forceinline__ float bf2f(unsigned short b) {
  union { unsigned u; float f; } x; x.u = (unsigned)b << 16;
  return x.f;
}

// ---------------------------------------------------------------------------
// TDM: DMA a [128 rows x 32 bf16] tile (row stride K elements) from global
// memory into LDS at lds_addr, padding each 64B row to 96B (LDS_STRIDE=48
// bf16) in hardware: pad_interval 16 DWORDs, pad_amount 8 DWORDs.
// D# layout per CDNA5 ISA 8.3/8.4. Tracked by TENSORcnt.
// ---------------------------------------------------------------------------
static __device__ __forceinline__ void tdm_load_tile_a(unsigned lds_addr,
                                                       const unsigned short* gaddr,
                                                       int K) {
  const unsigned long long ga = (unsigned long long)(uintptr_t)gaddr;
  u32x4 g0;
  g0[0] = 1u;                                   // count=1, user descriptor
  g0[1] = lds_addr;                             // LDS byte address
  g0[2] = (unsigned)(ga & 0xFFFFFFFFu);         // global_addr[31:0]
  g0[3] = (unsigned)((ga >> 32) & 0x01FFFFFFu)  // global_addr[56:32]
          | (2u << 30);                         // type = 2 ("image")
  i32x8 g1;
  g1[0] = (1 << 16)      // data_size = 1 -> 2 bytes
        | (1 << 20)      // pad_enable
        | (3 << 22)      // pad_interval: 16 DWORDs (one 32-elem bf16 row)
        | (7 << 25);     // pad_amount:    8 DWORDs (64B -> 96B row stride)
  g1[1] = (int)(32u << 16);    // tensor_dim0 = 32      (bits 79:48)
  g1[2] = (int)(128u << 16);   // tensor_dim1 = 128     (bits 111:80)
  g1[3] = (int)(32u << 16);    // tile_dim0   = 32      (bits 127:112)
  g1[4] = 128;                 // tile_dim1 = 128, tile_dim2 = 0
  g1[5] = K;                   // tensor_dim0_stride[31:0] (elements)
  g1[6] = 0;                   // stride[47:32]=0, dim1_stride lo = 0
  g1[7] = 0;
  const i32x4 gz = {0, 0, 0, 0};
#if __clang_major__ >= 23
  const i32x8 gz8 = {0, 0, 0, 0, 0, 0, 0, 0};
  __builtin_amdgcn_tensor_load_to_lds(g0, g1, gz, gz, gz8, 0);
#else
  __builtin_amdgcn_tensor_load_to_lds(g0, g1, gz, gz, 0);
#endif
}

// ---------------------------------------------------------------------------
// Kernel 1: fused pass over fp32 A (one block per row):
//   dis[i] = rsqrt( sum_j A[i,j] + 1 ) ;  Ab[i,j] = bf16( A[i,j] + (i==j) )
// ---------------------------------------------------------------------------
__global__ __launch_bounds__(256)
void rowsum_raw_kernel(const float* __restrict__ A,
                       unsigned short* __restrict__ Ab,
                       float* __restrict__ dis) {
  __shared__ float red[256];
  const int row = blockIdx.x;
  const float* arow = A + (size_t)row * NN;
  unsigned short* orow = Ab + (size_t)row * NN;
  float s = 0.f;
  for (int c = threadIdx.x * 4; c < NN; c += 1024) {
    float4 v = *(const float4*)(arow + c);
    s += (v.x + v.y) + (v.z + v.w);
    float t0 = v.x + ((c + 0) == row ? 1.f : 0.f);
    float t1 = v.y + ((c + 1) == row ? 1.f : 0.f);
    float t2 = v.z + ((c + 2) == row ? 1.f : 0.f);
    float t3 = v.w + ((c + 3) == row ? 1.f : 0.f);
    uint2 p;
    p.x = (unsigned)f2bf(t0) | ((unsigned)f2bf(t1) << 16);
    p.y = (unsigned)f2bf(t2) | ((unsigned)f2bf(t3) << 16);
    *(uint2*)(orow + c) = p;
  }
  red[threadIdx.x] = s;
  __syncthreads();
  for (int off = 128; off > 0; off >>= 1) {
    if (threadIdx.x < off) red[threadIdx.x] += red[threadIdx.x + off];
    __syncthreads();
  }
  if (threadIdx.x == 0) dis[row] = rsqrtf(red[0] + 1.0f);
}

// ---------------------------------------------------------------------------
// Kernel 2: in-place scale of the bf16 copy: Ab[i,j] *= dis[i]*dis[j]
// ---------------------------------------------------------------------------
__global__ __launch_bounds__(256)
void scale_bf16_kernel(unsigned short* __restrict__ Ab,
                       const float* __restrict__ dis) {
  const size_t base = ((size_t)blockIdx.x * 256 + threadIdx.x) * 8;
  const int row = (int)(base >> 13);       // / 8192
  const int col = (int)(base & (NN - 1));  // % 8192
  const float dr = dis[row];
  float4 d0 = *(const float4*)(dis + col);
  float4 d1 = *(const float4*)(dis + col + 4);
  union { uint4 q; unsigned short s[8]; } v;
  v.q = *(const uint4*)(Ab + base);
  const float dc[8] = {d0.x, d0.y, d0.z, d0.w, d1.x, d1.y, d1.z, d1.w};
  unsigned short o[8];
#pragma unroll
  for (int j = 0; j < 8; ++j) o[j] = f2bf(bf2f(v.s[j]) * dr * dc[j]);
  uint4 packed;
  packed.x = (unsigned)o[0] | ((unsigned)o[1] << 16);
  packed.y = (unsigned)o[2] | ((unsigned)o[3] << 16);
  packed.z = (unsigned)o[4] | ((unsigned)o[5] << 16);
  packed.w = (unsigned)o[6] | ((unsigned)o[7] << 16);
  *(uint4*)(Ab + base) = packed;
}

// ---------------------------------------------------------------------------
// Kernel 3: generic f32 -> bf16 (n must be a multiple of 4)
// ---------------------------------------------------------------------------
__global__ __launch_bounds__(256)
void cvt_bf16_kernel(const float* __restrict__ in, unsigned short* __restrict__ out, int n) {
  int i = (blockIdx.x * 256 + threadIdx.x) * 4;
  if (i >= n) return;
  float4 v = *(const float4*)(in + i);
  uint2 p;
  p.x = (unsigned)f2bf(v.x) | ((unsigned)f2bf(v.y) << 16);
  p.y = (unsigned)f2bf(v.z) | ((unsigned)f2bf(v.w) << 16);
  *(uint2*)(out + i) = p;
}

// ---------------------------------------------------------------------------
// Kernel 4: pad W3 [128,40] -> bf16 [128,64] (zeros), b3 [40] -> f32 [64]
// ---------------------------------------------------------------------------
__global__ __launch_bounds__(256)
void pad_w3_b3_kernel(const float* __restrict__ W3, const float* __restrict__ b3,
                      unsigned short* __restrict__ W3b, float* __restrict__ b3p) {
  int i = blockIdx.x * 256 + threadIdx.x;
  if (i < 128 * 64) {
    int r = i >> 6, c = i & 63;
    W3b[i] = f2bf(c < 40 ? W3[r * 40 + c] : 0.0f);
  }
  if (i < 64) b3p[i] = (i < 40) ? b3[i] : 0.0f;
}

// ---------------------------------------------------------------------------
// Kernel 5: WMMA GEMM  C[M,N] = A[M,K] @ B[K,N]  (+bias, +relu)
// A,B bf16 row-major; fp32 accumulate; output bf16 or f32.
// Block tile 128x64, K-step 32, 8 waves; wave w computes rows [w*16, w*16+16)
// across 4 n-tiles of 16. Double-buffered LDS; A tiles DMA'd by the TDM
// (one tensor_load_to_lds per K-step, wave 0 only, EXEC-independent);
// B tiles staged via regs + transpose scatter. Layouts per ISA 7.12.2.
// ---------------------------------------------------------------------------
#define BM 128
#define BN 64
#define BK 32
#define LDS_STRIDE 48  // bf16 units; 96B row stride (TDM pad), 16B aligned

template <bool RELU, bool BF16_OUT, bool HAS_BIAS>
__global__ __launch_bounds__(256)
void wmma_gemm_bf16_kernel(const unsigned short* __restrict__ A,
                           const unsigned short* __restrict__ B,
                           const float* __restrict__ bias,
                           void* __restrict__ Cout,
                           int M, int N, int K) {
  __shared__ unsigned short As[2][BM * LDS_STRIDE];  // 2 x 12 KB  [row][k]
  __shared__ unsigned short Bs[2][BN * LDS_STRIDE];  // 2 x  6 KB  [col][k] (transposed)

  const int tid   = threadIdx.x;
  const int wave  = tid >> 5;
  const int lane  = tid & 31;
  const int lhalf = lane >> 4;  // 0: lanes 0-15, 1: lanes 16-31
  const int l16   = lane & 15;

  const int m0 = blockIdx.x * BM;
  const int n0 = blockIdx.y * BN;

  const f32x8 zero = {0.f, 0.f, 0.f, 0.f, 0.f, 0.f, 0.f, 0.f};
  f32x8 acc[4];
#pragma unroll
  for (int i = 0; i < 4; ++i) acc[i] = zero;

  // A-tile DMA source: block-uniform row m0, column k0 (added per step)
  const unsigned short* atile = A + (size_t)m0 * K;
  // B-tile staging (transposed): thread t -> k row t>>3, 8 cols at (t&7)*8
  const int bk = tid >> 3;
  const int bc = (tid & 7) << 3;
  const unsigned short* bgp = B + (size_t)bk * N + n0 + bc;

  auto stage_a_tdm = [&](int buf, int k0) {
    if (wave == 0)
      tdm_load_tile_a((unsigned)(uintptr_t)&As[buf][0], atile + k0, K);
  };
  auto load_b = [&](int k0) -> uint4 {
    return *(const uint4*)(bgp + (size_t)k0 * N);
  };
  auto scatter_b = [&](int buf, uint4 q) {
    union { uint4 q; unsigned short s[8]; } v; v.q = q;
#pragma unroll
    for (int j = 0; j < 8; ++j)
      Bs[buf][(bc + j) * LDS_STRIDE + bk] = v.s[j];
  };

  // ---- prologue: stage K-tile 0 into buffer 0 ----
  stage_a_tdm(0, 0);
  scatter_b(0, load_b(0));
  if (wave == 0) __builtin_amdgcn_s_wait_tensorcnt(0);
  __syncthreads();

  int buf = 0;
  for (int k0 = 0; k0 < K; k0 += BK) {
    const bool has_next = (k0 + BK) < K;
    uint4 bnext;
    if (has_next) {
      stage_a_tdm(buf ^ 1, k0 + BK);  // TDM DMA -> other LDS buffer
      bnext = load_b(k0 + BK);        // global loads in flight during compute
    }

    // ---- A fragment: lane<16 row=wave*16+l16, K {0..7}u{16..23};
    //                  lane>=16 same row,       K {8..15}u{24..31} ----
    union { bf16x16 v; uint4 q[2]; } afrag;
    {
      const int arow = wave * 16 + l16;
      const int kb = lhalf * 8;
      afrag.q[0] = *(const uint4*)(&As[buf][arow * LDS_STRIDE + kb]);
      afrag.q[1] = *(const uint4*)(&As[buf][arow * LDS_STRIDE + kb + 16]);
    }
    // ---- B fragments (all four preloaded so ds_loads pipeline) ----
    union { bf16x16 v; uint4 q[2]; } bfrag[4];
#pragma unroll
    for (int nt = 0; nt < 4; ++nt) {
      const int bn = nt * 16 + l16;
      const int bkb = lhalf * 8;
      bfrag[nt].q[0] = *(const uint4*)(&Bs[buf][bn * LDS_STRIDE + bkb]);
      bfrag[nt].q[1] = *(const uint4*)(&Bs[buf][bn * LDS_STRIDE + bkb + 16]);
    }
    // keep all ds_loads above, all WMMAs below (single dscnt wait, 4 WMMAs
    // issue back-to-back)
    __builtin_amdgcn_sched_barrier(0);
#pragma unroll
    for (int nt = 0; nt < 4; ++nt) {
      acc[nt] = __builtin_amdgcn_wmma_f32_16x16x32_bf16(
          false, afrag.v, false, bfrag[nt].v, (short)0, acc[nt], false, false);
    }

    if (has_next) scatter_b(buf ^ 1, bnext);  // waits loadcnt, writes other buf
    if (wave == 0) __builtin_amdgcn_s_wait_tensorcnt(0);
    __syncthreads();
    buf ^= 1;
  }

  // ---- epilogue: C/D layout — VGPR r holds row (lhalf*8 + r), col = l16 ----
  const int row_base = m0 + wave * 16 + lhalf * 8;
#pragma unroll
  for (int nt = 0; nt < 4; ++nt) {
    const int col = n0 + nt * 16 + l16;
    float bv = 0.0f;
    if constexpr (HAS_BIAS) bv = bias[col];
#pragma unroll
    for (int r = 0; r < 8; ++r) {
      float v = acc[nt][r];
      if constexpr (HAS_BIAS) v += bv;
      if constexpr (RELU) v = fmaxf(v, 0.0f);
      const size_t idx = (size_t)(row_base + r) * N + col;
      if constexpr (BF16_OUT) ((unsigned short*)Cout)[idx] = f2bf(v);
      else                    ((float*)Cout)[idx] = v;
    }
  }
}

// ---------------------------------------------------------------------------
// Kernel 6: row-wise log_softmax over the first 40 of 64 padded columns
// ---------------------------------------------------------------------------
__global__ __launch_bounds__(64)
void log_softmax_kernel(const float* __restrict__ logits, float* __restrict__ out) {
  __shared__ float red[64];
  const int row = blockIdx.x;
  const int t = threadIdx.x;
  const float v = (t < 40) ? logits[row * 64 + t] : -3.4e38f;
  red[t] = v;
  __syncthreads();
  for (int off = 32; off > 0; off >>= 1) {
    if (t < off) red[t] = fmaxf(red[t], red[t + off]);
    __syncthreads();
  }
  const float m = red[0];
  __syncthreads();
  red[t] = (t < 40) ? __expf(v - m) : 0.0f;
  __syncthreads();
  for (int off = 32; off > 0; off >>= 1) {
    if (t < off) red[t] += red[t + off];
    __syncthreads();
  }
  const float lse = m + __logf(red[0]);
  if (t < 40) out[row * 40 + t] = v - lse;
}

// ---------------------------------------------------------------------------
// Host orchestration
// ---------------------------------------------------------------------------
extern "C" void kernel_launch(void* const* d_in, const int* in_sizes, int n_in,
                              void* d_out, int out_size, void* d_ws, size_t ws_size,
                              hipStream_t stream) {
  (void)in_sizes; (void)n_in; (void)out_size; (void)ws_size;
  const float* X  = (const float*)d_in[0];  // [8192, 512]
  const float* A  = (const float*)d_in[1];  // [8192, 8192]
  const float* W1 = (const float*)d_in[2];  // [512, 256]
  const float* b1 = (const float*)d_in[3];  // [256]
  const float* W2 = (const float*)d_in[4];  // [256, 128]
  const float* b2 = (const float*)d_in[5];  // [128]
  const float* W3 = (const float*)d_in[6];  // [128, 40]
  const float* b3 = (const float*)d_in[7];  // [40]
  float* out = (float*)d_out;               // [8192, 40]

  // workspace carve-up (all sizes 16B-aligned)
  char* ws = (char*)d_ws;
  unsigned short* Ab  = (unsigned short*)ws; ws += (size_t)NN * NN * 2;   // 128 MB (fits L2)
  unsigned short* Xb  = (unsigned short*)ws; ws += (size_t)NN * 512 * 2;  //   8 MB
  unsigned short* P   = (unsigned short*)ws; ws += (size_t)NN * 256 * 2;  //   4 MB (reused)
  unsigned short* H1  = (unsigned short*)ws; ws += (size_t)NN * 256 * 2;  //   4 MB
  unsigned short* H2  = (unsigned short*)ws; ws += (size_t)NN * 128 * 2;  //   2 MB
  float*          Lg  = (float*)ws;          ws += (size_t)NN * 64 * 4;   //   2 MB
  unsigned short* W1b = (unsigned short*)ws; ws += 512 * 256 * 2;
  unsigned short* W2b = (unsigned short*)ws; ws += 256 * 128 * 2;
  unsigned short* W3b = (unsigned short*)ws; ws += 128 * 64 * 2;
  float*          b3p = (float*)ws;          ws += 64 * 4;
  float*          dis = (float*)ws;          ws += NN * 4;

  // 1) single fp32 pass over A: row sums (-> dis) + raw bf16(A+I)
  rowsum_raw_kernel<<<NN, 256, 0, stream>>>(A, Ab, dis);
  // 2) in-place D^-1/2 (.) D^-1/2 scaling of the bf16 copy (L2 traffic)
  scale_bf16_kernel<<<(unsigned)((NN * (size_t)NN) / (256 * 8)), 256, 0, stream>>>(Ab, dis);
  // 3) precision conversions
  cvt_bf16_kernel<<<(NN * 512 / 4 + 255) / 256, 256, 0, stream>>>(X, Xb, NN * 512);
  cvt_bf16_kernel<<<(512 * 256 / 4 + 255) / 256, 256, 0, stream>>>(W1, W1b, 512 * 256);
  cvt_bf16_kernel<<<(256 * 128 / 4 + 255) / 256, 256, 0, stream>>>(W2, W2b, 256 * 128);
  pad_w3_b3_kernel<<<(128 * 64 + 255) / 256, 256, 0, stream>>>(W3, b3, W3b, b3p);

  // Layer 1: P = X @ W1 ; H1 = relu(A_hat @ P + b1)
  wmma_gemm_bf16_kernel<false, true, false>
      <<<dim3(NN / BM, 256 / BN), 256, 0, stream>>>(Xb, W1b, nullptr, P, NN, 256, 512);
  wmma_gemm_bf16_kernel<true, true, true>
      <<<dim3(NN / BM, 256 / BN), 256, 0, stream>>>(Ab, P, b1, H1, NN, 256, NN);

  // Layer 2: P = H1 @ W2 ; H2 = relu(A_hat @ P + b2)
  wmma_gemm_bf16_kernel<false, true, false>
      <<<dim3(NN / BM, 128 / BN), 256, 0, stream>>>(H1, W2b, nullptr, P, NN, 128, 256);
  wmma_gemm_bf16_kernel<true, true, true>
      <<<dim3(NN / BM, 128 / BN), 256, 0, stream>>>(Ab, P, b2, H2, NN, 128, NN);

  // Layer 3 (padded to 64 cols): P = H2 @ W3b ; logits = A_hat @ P + b3
  wmma_gemm_bf16_kernel<false, true, false>
      <<<dim3(NN / BM, 64 / BN), 256, 0, stream>>>(H2, W3b, nullptr, P, NN, 64, 128);
  wmma_gemm_bf16_kernel<false, false, true>
      <<<dim3(NN / BM, 64 / BN), 256, 0, stream>>>(Ab, P, b3p, Lg, NN, 64, NN);

  // 4) log_softmax over 40 valid columns
  log_softmax_kernel<<<NN, 64, 0, stream>>>(Lg, out);
}
